// TiTokEncoder_34943853920926
// MI455X (gfx1250) — compile-verified, matrix operand order checked by reading, not run on hardware
//
#include <hip/hip_runtime.h>
#include <math.h>

// ---------------------------------------------------------------------------
// Types for CDNA5 WMMA
// ---------------------------------------------------------------------------
typedef __attribute__((ext_vector_type(16))) __bf16 v16bf;
typedef __attribute__((ext_vector_type(8)))  __bf16 v8bf;
typedef __attribute__((ext_vector_type(8)))  float  v8f;

#define WMMA_BF16(a, b, c) \
    __builtin_amdgcn_wmma_f32_16x16x32_bf16(false, (a), false, (b), (short)0, (c), false, false)

// Model constants
#define Bsz 8
#define Cch 3
#define Tdim 16
#define HD 128
#define WD 128
#define NT 4
#define NH 16
#define NW 16
#define NPATCH 1024           // NT*NH*NW per batch
#define PIN 768               // C*PT*PH*PW
#define WIDTH 256
#define HEADS 4
#define HEAD_DIM 64
#define MLP_DIM 1024
#define NLAYERS 4
#define LTOK 256
#define SEQ 1280              // NPATCH + LTOK
#define NTOK (Bsz * SEQ)      // 10240
#define PAIRS 32
#define OUTD 5

// ---------------------------------------------------------------------------
// DPP cross-lane reductions over each 16-lane row (no LDS, no DScnt waits).
//   xor1 = quad_perm[1,0,3,2] (0xB1), xor2 = quad_perm[2,3,0,1] (0x4E),
//   xor7 = row_half_mirror (0x141),   xor15 = row_mirror (0x140).
// After {xor1, xor2, xor7, xor15} every lane of a 16-lane row holds the
// row-wide reduction; the two rows (lanes 0-15 / 16-31) stay independent,
// matching the WMMA C-layout half-wave row groups.
// ---------------------------------------------------------------------------
template <int CTRL>
__device__ __forceinline__ float dpp_mov_f32(float v) {
    int t = __builtin_amdgcn_update_dpp(0, __float_as_int(v), CTRL, 0xF, 0xF, true);
    return __int_as_float(t);
}
__device__ __forceinline__ float red16_max(float v) {
    v = fmaxf(v, dpp_mov_f32<0xB1>(v));
    v = fmaxf(v, dpp_mov_f32<0x4E>(v));
    v = fmaxf(v, dpp_mov_f32<0x141>(v));
    v = fmaxf(v, dpp_mov_f32<0x140>(v));
    return v;
}
__device__ __forceinline__ float red16_sum(float v) {
    v += dpp_mov_f32<0xB1>(v);
    v += dpp_mov_f32<0x4E>(v);
    v += dpp_mov_f32<0x141>(v);
    v += dpp_mov_f32<0x140>(v);
    return v;
}

// ---------------------------------------------------------------------------
// Fragment loader: 16-bit A/B fragment (16 rows x 32 K), ISA layout:
//   lane l: row = l&15, half = l>>4; elements K in
//   {half*8 .. half*8+7} U {16+half*8 .. 16+half*8+7}  (two 16B chunks)
// ---------------------------------------------------------------------------
__device__ __forceinline__ v16bf load_frag(const __bf16* base, int ld, int lane) {
    int m = lane & 15, half = lane >> 4;
    const __bf16* p = base + (size_t)m * ld + half * 8;
    v8bf lo = *(const v8bf*)(p);
    v8bf hi = *(const v8bf*)(p + 16);
    return __builtin_shufflevector(lo, hi, 0, 1, 2, 3, 4, 5, 6, 7,
                                           8, 9, 10, 11, 12, 13, 14, 15);
}

// ---------------------------------------------------------------------------
// Spectral norm power iteration (5 steps), one block per matrix.
// ---------------------------------------------------------------------------
__global__ void sigma_kernel(const float* __restrict__ W, int M, int N, float* out) {
    __shared__ float u[256];
    __shared__ float v[768];
    __shared__ float red[256];
    int t = threadIdx.x;
    for (int i = t; i < M; i += 256) u[i] = rsqrtf((float)M);
    __syncthreads();
    for (int it = 0; it < 5; ++it) {
        // v = W^T u
        for (int j = t; j < N; j += 256) {
            float s = 0.f;
            for (int i = 0; i < M; ++i) s += W[(size_t)i * N + j] * u[i];
            v[j] = s;
        }
        __syncthreads();
        float s = 0.f;
        for (int j = t; j < N; j += 256) s += v[j] * v[j];
        red[t] = s; __syncthreads();
        for (int o = 128; o > 0; o >>= 1) { if (t < o) red[t] += red[t + o]; __syncthreads(); }
        float nv = sqrtf(red[0]) + 1e-12f;
        __syncthreads();
        for (int j = t; j < N; j += 256) v[j] /= nv;
        __syncthreads();
        // u = W v
        for (int i = t; i < M; i += 256) {
            float s2 = 0.f;
            for (int j = 0; j < N; ++j) s2 += W[(size_t)i * N + j] * v[j];
            u[i] = s2;
        }
        __syncthreads();
        float s3 = 0.f;
        for (int i = t; i < M; i += 256) s3 += u[i] * u[i];
        red[t] = s3; __syncthreads();
        for (int o = 128; o > 0; o >>= 1) { if (t < o) red[t] += red[t + o]; __syncthreads(); }
        float nu = sqrtf(red[0]) + 1e-12f;
        __syncthreads();
        for (int i = t; i < M; i += 256) u[i] /= nu;
        __syncthreads();
    }
    // sigma = u^T W v
    float s = 0.f;
    for (int idx = t; idx < M * N; idx += 256) {
        int i = idx / N, j = idx % N;
        s += u[i] * W[idx] * v[j];
    }
    red[t] = s; __syncthreads();
    for (int o = 128; o > 0; o >>= 1) { if (t < o) red[t] += red[t + o]; __syncthreads(); }
    if (t == 0) *out = red[0];
}

// ---------------------------------------------------------------------------
// fp32 -> bf16 weight convert (optional divide by sigma read on device)
// ---------------------------------------------------------------------------
__global__ void cvt_bf16_kernel(const float* __restrict__ in, __bf16* __restrict__ out,
                                int n, const float* sigma) {
    int i = blockIdx.x * 256 + threadIdx.x;
    if (i >= n) return;
    float sc = sigma ? (1.0f / (*sigma)) : 1.0f;
    out[i] = (__bf16)(in[i] * sc);
}

// ---------------------------------------------------------------------------
// Patchify x (B,C,T,H,W) -> pbf bf16 [B*1024, 768]
// ---------------------------------------------------------------------------
__global__ void patchify_kernel(const float* __restrict__ x, __bf16* __restrict__ out) {
    int idx = blockIdx.x * 256 + threadIdx.x;
    if (idx >= Bsz * NPATCH * PIN) return;
    int k = idx % PIN;
    int row = idx / PIN;
    int b = row / NPATCH, tok = row % NPATCH;
    int tz = tok / (NH * NW), ty = (tok / NW) % NH, tx = tok % NW;
    int c = k / 256, r = k % 256;
    int pt = r / 64, ph = (r / 8) % 8, pw = r % 8;
    size_t off = (((size_t)b * Cch + c) * Tdim + (tz * 4 + pt)) * (HD * WD)
               + (size_t)(ty * 8 + ph) * WD + (tx * 8 + pw);
    out[idx] = (__bf16)x[off];
}

// ---------------------------------------------------------------------------
// RoPE tables [1280, 32]
// ---------------------------------------------------------------------------
__global__ void rope_table_kernel(float* __restrict__ cosb, float* __restrict__ sinb) {
    int idx = blockIdx.x * 256 + threadIdx.x;
    if (idx >= SEQ * PAIRS) return;
    int p = idx % PAIRS, pos = idx / PAIRS;
    float ang = 0.f;
    if (pos < NPATCH) {
        int tz = pos / (NH * NW), ty = (pos / NW) % NH, tx = pos % NW;
        if (p < 11)      ang = (float)tz * powf(10000.f, -(float)p / 11.f);
        else if (p < 22) ang = (float)ty * powf(10000.f, -(float)(p - 11) / 11.f);
        else             ang = (float)tx * powf(10000.f, -(float)(p - 22) / 10.f);
    }
    cosb[idx] = cosf(ang);
    sinb[idx] = sinf(ang);
}

// ---------------------------------------------------------------------------
// Fill latent rows of h with mask token
// ---------------------------------------------------------------------------
__global__ void init_latent_kernel(float* __restrict__ h, const float* __restrict__ mask) {
    int idx = blockIdx.x * 256 + threadIdx.x;
    if (idx >= Bsz * LTOK * WIDTH) return;
    int b = idx / (LTOK * WIDTH);
    int rem = idx % (LTOK * WIDTH);
    h[((size_t)b * SEQ + NPATCH) * WIDTH + rem] = mask[0];
}

// ---------------------------------------------------------------------------
// Generic bf16 WMMA GEMM: D = A[M,K] @ Bw[N,K]^T + bias, epilogue variants.
// One wave -> 16x64 output tile; 8 waves (256 thr) per block.
// Output row mapping: orow = (row / rpc) * cstride + (row % rpc)
// ---------------------------------------------------------------------------
#define EP_F32   0
#define EP_BF16  1
#define EP_GELU  2
#define EP_RESID 3

template <int EPI>
__global__ void gemm_wmma_kernel(const __bf16* __restrict__ A, int lda,
                                 const __bf16* __restrict__ Bw, int ldb,
                                 const float* __restrict__ bias,
                                 float* __restrict__ outF, __bf16* __restrict__ outH,
                                 int M, int N, int K,
                                 int rpc, int cstride, int out_ld) {
    int wave = threadIdx.x >> 5;
    int lane = threadIdx.x & 31;
    int mtile = blockIdx.x * (blockDim.x >> 5) + wave;
    if (mtile * 16 >= M) return;
    int nbase = blockIdx.y * 64;

    v8f acc0 = {}, acc1 = {}, acc2 = {}, acc3 = {};
    const __bf16* arow = A + (size_t)(mtile * 16) * lda;
    const __bf16* b0 = Bw + (size_t)(nbase +  0) * ldb;
    const __bf16* b1 = Bw + (size_t)(nbase + 16) * ldb;
    const __bf16* b2 = Bw + (size_t)(nbase + 32) * ldb;
    const __bf16* b3 = Bw + (size_t)(nbase + 48) * ldb;

    for (int k0 = 0; k0 < K; k0 += 32) {
        v16bf a  = load_frag(arow + k0, lda, lane);
        v16bf f0 = load_frag(b0 + k0, ldb, lane);
        v16bf f1 = load_frag(b1 + k0, ldb, lane);
        v16bf f2 = load_frag(b2 + k0, ldb, lane);
        v16bf f3 = load_frag(b3 + k0, ldb, lane);
        if (k0 + 32 < K) {
            // gfx1250 global_prefetch_b8: pull next K-step tiles toward L0/L2
            __builtin_prefetch(arow + k0 + 32, 0, 1);
            __builtin_prefetch(b0 + k0 + 32, 0, 1);
            __builtin_prefetch(b1 + k0 + 32, 0, 1);
            __builtin_prefetch(b2 + k0 + 32, 0, 1);
            __builtin_prefetch(b3 + k0 + 32, 0, 1);
        }
        acc0 = WMMA_BF16(a, f0, acc0);
        acc1 = WMMA_BF16(a, f1, acc1);
        acc2 = WMMA_BF16(a, f2, acc2);
        acc3 = WMMA_BF16(a, f3, acc3);
    }

    int n = lane & 15, mh = lane >> 4;
    v8f accs[4] = {acc0, acc1, acc2, acc3};
#pragma unroll
    for (int j = 0; j < 4; ++j) {
        int col = nbase + j * 16 + n;
        float bv = bias ? bias[col] : 0.f;
#pragma unroll
        for (int r = 0; r < 8; ++r) {
            int row = mtile * 16 + mh * 8 + r;
            int orow = (row / rpc) * cstride + (row % rpc);
            float v = accs[j][r] + bv;
            if (EPI == EP_GELU) v = 0.5f * v * (1.f + erff(v * 0.70710678118654752f));
            if (EPI == EP_RESID)      outF[(size_t)orow * out_ld + col] += v;
            else if (EPI == EP_F32)   outF[(size_t)orow * out_ld + col] = v;
            else                      outH[(size_t)orow * out_ld + col] = (__bf16)v;
        }
    }
}

// ---------------------------------------------------------------------------
// LayerNorm fp32 row -> bf16 row. One block (256 thr) per token.
// ---------------------------------------------------------------------------
__global__ void layernorm_kernel(const float* __restrict__ h, const float* __restrict__ w,
                                 const float* __restrict__ b, __bf16* __restrict__ out) {
    int row = blockIdx.x, t = threadIdx.x;
    __shared__ float red[256];
    float x = h[(size_t)row * WIDTH + t];
    red[t] = x; __syncthreads();
    for (int o = 128; o > 0; o >>= 1) { if (t < o) red[t] += red[t + o]; __syncthreads(); }
    float mu = red[0] * (1.f / WIDTH);
    __syncthreads();
    float d = x - mu;
    red[t] = d * d; __syncthreads();
    for (int o = 128; o > 0; o >>= 1) { if (t < o) red[t] += red[t + o]; __syncthreads(); }
    float var = red[0] * (1.f / WIDTH);
    float y = d * rsqrtf(var + 1e-5f) * w[t] + b[t];
    out[(size_t)row * WIDTH + t] = (__bf16)y;
}

// ---------------------------------------------------------------------------
// RoPE + pack: qkv fp32 [B*SEQ, 768] -> qb,kb bf16 [bh*SEQ + s, 64],
//                                      vbT bf16 [bh*64 + d, SEQ] (transposed)
// ---------------------------------------------------------------------------
__global__ void rope_pack_kernel(const float* __restrict__ qkv,
                                 const float* __restrict__ cosb, const float* __restrict__ sinb,
                                 __bf16* __restrict__ qb, __bf16* __restrict__ kb,
                                 __bf16* __restrict__ vbT) {
    int idx = blockIdx.x * 256 + threadIdx.x;     // B*SEQ*HEADS*PAIRS
    if (idx >= Bsz * SEQ * HEADS * PAIRS) return;
    int p = idx % PAIRS;
    int h = (idx / PAIRS) % HEADS;
    int s = (idx / (PAIRS * HEADS)) % SEQ;
    int b = idx / (PAIRS * HEADS * SEQ);
    size_t row = (size_t)(b * SEQ + s) * (3 * WIDTH);
    int d0 = h * HEAD_DIM + 2 * p;
    float qa = qkv[row + d0],             qbv = qkv[row + d0 + 1];
    float ka = qkv[row + WIDTH + d0],     kbv = qkv[row + WIDTH + d0 + 1];
    float va = qkv[row + 2 * WIDTH + d0], vb2 = qkv[row + 2 * WIDTH + d0 + 1];
    float c = cosb[s * PAIRS + p], sn = sinb[s * PAIRS + p];
    size_t orow = ((size_t)(b * HEADS + h) * SEQ + s) * HEAD_DIM + 2 * p;
    qb[orow]     = (__bf16)(qa * c - qbv * sn);
    qb[orow + 1] = (__bf16)(qa * sn + qbv * c);
    kb[orow]     = (__bf16)(ka * c - kbv * sn);
    kb[orow + 1] = (__bf16)(ka * sn + kbv * c);
    size_t vcol = ((size_t)(b * HEADS + h) * HEAD_DIM + 2 * p) * SEQ + s;
    vbT[vcol]       = (__bf16)va;
    vbT[vcol + SEQ] = (__bf16)vb2;
}

// ---------------------------------------------------------------------------
// Flash attention: one wave per (bh, 16-query tile). Online softmax with DPP
// row reductions; WMMA for Q@K^T and P@V. P re-layout (C-format -> A-frag)
// bounced via 1KB LDS.
// ---------------------------------------------------------------------------
__global__ void attention_kernel(const __bf16* __restrict__ qb, const __bf16* __restrict__ kb,
                                 const __bf16* __restrict__ vbT, __bf16* __restrict__ obf) {
    int bh = blockIdx.y;     // 0..31
    int qt = blockIdx.x;     // 0..79
    int lane = threadIdx.x;
    const float scale = 0.125f;   // HEAD_DIM^-0.5
    __shared__ __bf16 plds[16 * 32];

    const __bf16* qbase = qb + ((size_t)bh * SEQ + qt * 16) * HEAD_DIM;
    v16bf qa0 = load_frag(qbase, HEAD_DIM, lane);        // K dims 0..31
    v16bf qa1 = load_frag(qbase + 32, HEAD_DIM, lane);   // K dims 32..63

    v8f o0 = {}, o1 = {}, o2 = {}, o3 = {};
    float mrow[8], lrow[8];
#pragma unroll
    for (int r = 0; r < 8; ++r) { mrow[r] = -1e30f; lrow[r] = 0.f; }
    int n = lane & 15, mh = lane >> 4;

    for (int c0 = 0; c0 < SEQ; c0 += 32) {
        // ---- scores: two 16x16 tiles, each K=64 over head dim ----
        const __bf16* kb0 = kb + ((size_t)bh * SEQ + c0) * HEAD_DIM;
        const __bf16* kb1 = kb + ((size_t)bh * SEQ + c0 + 16) * HEAD_DIM;
        v16bf k00 = load_frag(kb0,      HEAD_DIM, lane);
        v16bf k01 = load_frag(kb0 + 32, HEAD_DIM, lane);
        v16bf k10 = load_frag(kb1,      HEAD_DIM, lane);
        v16bf k11 = load_frag(kb1 + 32, HEAD_DIM, lane);
        v8f s0 = {}, s1 = {};
        s0 = WMMA_BF16(qa0, k00, s0); s0 = WMMA_BF16(qa1, k01, s0);
        s1 = WMMA_BF16(qa0, k10, s1); s1 = WMMA_BF16(qa1, k11, s1);

        // ---- online softmax (rows mh*8+r; DPP reduce over 16 lanes) ----
        float nm[8], p0[8], p1[8], alpha[8];
#pragma unroll
        for (int r = 0; r < 8; ++r) {
            float v = red16_max(fmaxf(s0[r], s1[r]) * scale);
            nm[r] = fmaxf(mrow[r], v);
        }
#pragma unroll
        for (int r = 0; r < 8; ++r) {
            alpha[r] = __expf(mrow[r] - nm[r]);
            p0[r] = __expf(s0[r] * scale - nm[r]);
            p1[r] = __expf(s1[r] * scale - nm[r]);
            float sum = red16_sum(p0[r] + p1[r]);
            lrow[r] = lrow[r] * alpha[r] + sum;
            mrow[r] = nm[r];
        }
#pragma unroll
        for (int r = 0; r < 8; ++r) {
            o0[r] *= alpha[r]; o1[r] *= alpha[r];
            o2[r] *= alpha[r]; o3[r] *= alpha[r];
        }

        // ---- re-layout P (C format) -> row-major [16][32] in LDS ----
        __syncthreads();
#pragma unroll
        for (int r = 0; r < 8; ++r) {
            plds[(mh * 8 + r) * 32 + n]      = (__bf16)p0[r];
            plds[(mh * 8 + r) * 32 + 16 + n] = (__bf16)p1[r];
        }
        __syncthreads();
        v16bf pa = load_frag(plds, 32, lane);

        // ---- O += P @ V : V fragments from transposed V (contiguous) ----
        const __bf16* vb = vbT + (size_t)bh * HEAD_DIM * SEQ + c0;
        v16bf v0 = load_frag(vb,                       SEQ, lane);
        v16bf v1 = load_frag(vb + (size_t)16 * SEQ,    SEQ, lane);
        v16bf v2 = load_frag(vb + (size_t)32 * SEQ,    SEQ, lane);
        v16bf v3 = load_frag(vb + (size_t)48 * SEQ,    SEQ, lane);
        o0 = WMMA_BF16(pa, v0, o0);
        o1 = WMMA_BF16(pa, v1, o1);
        o2 = WMMA_BF16(pa, v2, o2);
        o3 = WMMA_BF16(pa, v3, o3);
    }

    // ---- normalize and store merged heads: obf[b*SEQ + s, h*64 + d] ----
    int b = bh >> 2, h = bh & 3;
    float inv[8];
#pragma unroll
    for (int r = 0; r < 8; ++r) inv[r] = 1.f / lrow[r];
    v8f os[4] = {o0, o1, o2, o3};
#pragma unroll
    for (int j = 0; j < 4; ++j) {
#pragma unroll
        for (int r = 0; r < 8; ++r) {
            int srow = qt * 16 + mh * 8 + r;
            int d = h * HEAD_DIM + j * 16 + n;
            obf[((size_t)b * SEQ + srow) * WIDTH + d] = (__bf16)(os[j][r] * inv[r]);
        }
    }
}

// ---------------------------------------------------------------------------
// Final: LN(latent) @ (Wout/sigma)^T + bout -> out (B,256,5) fp32
// ---------------------------------------------------------------------------
__global__ void final_kernel(const float* __restrict__ h, const float* __restrict__ lnw,
                             const float* __restrict__ lnb, const float* __restrict__ Wout,
                             const float* __restrict__ bout, const float* __restrict__ sigma,
                             float* __restrict__ out) {
    int row = blockIdx.x;            // b*256 + l
    int b = row >> 8, l = row & 255;
    int t = threadIdx.x;
    __shared__ float red[256];
    __shared__ float y[256];
    const float* hrow = h + ((size_t)b * SEQ + NPATCH + l) * WIDTH;
    float x = hrow[t];
    red[t] = x; __syncthreads();
    for (int o = 128; o > 0; o >>= 1) { if (t < o) red[t] += red[t + o]; __syncthreads(); }
    float mu = red[0] * (1.f / WIDTH);
    __syncthreads();
    float d = x - mu;
    red[t] = d * d; __syncthreads();
    for (int o = 128; o > 0; o >>= 1) { if (t < o) red[t] += red[t + o]; __syncthreads(); }
    float var = red[0] * (1.f / WIDTH);
    __syncthreads();
    y[t] = d * rsqrtf(var + 1e-5f) * lnw[t] + lnb[t];
    __syncthreads();
    float sc = 1.f / (*sigma);
    for (int o5 = 0; o5 < OUTD; ++o5) {
        float s = y[t] * Wout[o5 * WIDTH + t] * sc;
        red[t] = s; __syncthreads();
        for (int o = 128; o > 0; o >>= 1) { if (t < o) red[t] += red[t + o]; __syncthreads(); }
        if (t == 0) out[(size_t)row * OUTD + o5] = red[0] + bout[o5];
        __syncthreads();
    }
}

// ---------------------------------------------------------------------------
// Host orchestration
// ---------------------------------------------------------------------------
extern "C" void kernel_launch(void* const* d_in, const int* in_sizes, int n_in,
                              void* d_out, int out_size, void* d_ws, size_t ws_size,
                              hipStream_t stream) {
    const float* x          = (const float*)d_in[0];
    const float* mask_token = (const float*)d_in[2];
    const float* proj_in_w  = (const float*)d_in[3];
    const float* proj_in_b  = (const float*)d_in[4];
    const float* qkv_w      = (const float*)d_in[5];
    const float* qkv_b      = (const float*)d_in[6];
    const float* attn_o_w   = (const float*)d_in[7];
    const float* attn_o_b   = (const float*)d_in[8];
    const float* ln1_w      = (const float*)d_in[9];
    const float* ln1_b      = (const float*)d_in[10];
    const float* ln2_w      = (const float*)d_in[11];
    const float* ln2_b      = (const float*)d_in[12];
    const float* mlp1_w     = (const float*)d_in[13];
    const float* mlp1_b     = (const float*)d_in[14];
    const float* mlp2_w     = (const float*)d_in[15];
    const float* mlp2_b     = (const float*)d_in[16];
    const float* ln_post_w  = (const float*)d_in[17];
    const float* ln_post_b  = (const float*)d_in[18];
    const float* proj_out_w = (const float*)d_in[19];
    const float* proj_out_b = (const float*)d_in[20];

    // workspace carving (256B aligned)
    char* ws = (char*)d_ws;
    size_t off = 0;
    auto carve = [&](size_t bytes) -> void* {
        void* p = ws + off;
        off = (off + bytes + 255) & ~(size_t)255;
        return p;
    };
    float*  sigma_in  = (float*)carve(2 * sizeof(float));
    float*  sigma_out = sigma_in + 1;
    float*  hbuf  = (float*)carve((size_t)NTOK * WIDTH * 4);
    __bf16* pbf   = (__bf16*)carve((size_t)Bsz * NPATCH * PIN * 2);
    __bf16* w_in  = (__bf16*)carve((size_t)WIDTH * PIN * 2);
    __bf16* wqkv  = (__bf16*)carve((size_t)NLAYERS * 3 * WIDTH * WIDTH * 2);
    __bf16* wo    = (__bf16*)carve((size_t)NLAYERS * WIDTH * WIDTH * 2);
    __bf16* w1    = (__bf16*)carve((size_t)NLAYERS * MLP_DIM * WIDTH * 2);
    __bf16* w2    = (__bf16*)carve((size_t)NLAYERS * WIDTH * MLP_DIM * 2);
    float*  cosb  = (float*)carve((size_t)SEQ * PAIRS * 4);
    float*  sinb  = (float*)carve((size_t)SEQ * PAIRS * 4);
    __bf16* ybf   = (__bf16*)carve((size_t)NTOK * WIDTH * 2);
    float*  qkvf  = (float*)carve((size_t)NTOK * 3 * WIDTH * 4);
    __bf16* qb    = (__bf16*)carve((size_t)NTOK * WIDTH * 2);
    __bf16* kb    = (__bf16*)carve((size_t)NTOK * WIDTH * 2);
    __bf16* vbT   = (__bf16*)carve((size_t)NTOK * WIDTH * 2);
    __bf16* obf   = (__bf16*)carve((size_t)NTOK * WIDTH * 2);
    __bf16* gbf   = (__bf16*)carve((size_t)NTOK * MLP_DIM * 2);

    // 1) spectral sigmas
    sigma_kernel<<<1, 256, 0, stream>>>(proj_in_w, WIDTH, PIN, sigma_in);
    sigma_kernel<<<1, 256, 0, stream>>>(proj_out_w, OUTD, WIDTH, sigma_out);

    // 2) weight conversion to bf16 (proj_in pre-scaled by 1/sigma)
    auto cvt = [&](const float* src, __bf16* dst, int n, const float* sg) {
        cvt_bf16_kernel<<<(n + 255) / 256, 256, 0, stream>>>(src, dst, n, sg);
    };
    cvt(proj_in_w, w_in, WIDTH * PIN, sigma_in);
    cvt(qkv_w,    wqkv, NLAYERS * 3 * WIDTH * WIDTH, nullptr);
    cvt(attn_o_w, wo,   NLAYERS * WIDTH * WIDTH, nullptr);
    cvt(mlp1_w,   w1,   NLAYERS * MLP_DIM * WIDTH, nullptr);
    cvt(mlp2_w,   w2,   NLAYERS * WIDTH * MLP_DIM, nullptr);

    // 3) patchify, rope tables, latent init
    {
        int n = Bsz * NPATCH * PIN;
        patchify_kernel<<<(n + 255) / 256, 256, 0, stream>>>(x, pbf);
    }
    rope_table_kernel<<<(SEQ * PAIRS + 255) / 256, 256, 0, stream>>>(cosb, sinb);
    {
        int n = Bsz * LTOK * WIDTH;
        init_latent_kernel<<<(n + 255) / 256, 256, 0, stream>>>(hbuf, mask_token);
    }

    // 4) proj_in: [8192,768] @ w_in^T -> hbuf patch rows (chunked row map)
    {
        int M = Bsz * NPATCH;
        dim3 grid(M / 16 / 8, WIDTH / 64);
        gemm_wmma_kernel<EP_F32><<<grid, 256, 0, stream>>>(
            pbf, PIN, w_in, PIN, proj_in_b, hbuf, nullptr,
            M, WIDTH, PIN, NPATCH, SEQ, WIDTH);
    }

    // 5) transformer layers
    for (int i = 0; i < NLAYERS; ++i) {
        layernorm_kernel<<<NTOK, 256, 0, stream>>>(hbuf, ln1_w + i * WIDTH, ln1_b + i * WIDTH, ybf);

        dim3 gq(NTOK / 16 / 8, 3 * WIDTH / 64);
        gemm_wmma_kernel<EP_F32><<<gq, 256, 0, stream>>>(
            ybf, WIDTH, wqkv + (size_t)i * 3 * WIDTH * WIDTH, WIDTH,
            qkv_b + i * 3 * WIDTH, qkvf, nullptr,
            NTOK, 3 * WIDTH, WIDTH, NTOK, NTOK, 3 * WIDTH);

        {
            int n = Bsz * SEQ * HEADS * PAIRS;
            rope_pack_kernel<<<(n + 255) / 256, 256, 0, stream>>>(qkvf, cosb, sinb, qb, kb, vbT);
        }

        attention_kernel<<<dim3(SEQ / 16, Bsz * HEADS), 32, 0, stream>>>(qb, kb, vbT, obf);

        dim3 go(NTOK / 16 / 8, WIDTH / 64);
        gemm_wmma_kernel<EP_RESID><<<go, 256, 0, stream>>>(
            obf, WIDTH, wo + (size_t)i * WIDTH * WIDTH, WIDTH,
            attn_o_b + i * WIDTH, hbuf, nullptr,
            NTOK, WIDTH, WIDTH, NTOK, NTOK, WIDTH);

        layernorm_kernel<<<NTOK, 256, 0, stream>>>(hbuf, ln2_w + i * WIDTH, ln2_b + i * WIDTH, ybf);

        dim3 g1(NTOK / 16 / 8, MLP_DIM / 64);
        gemm_wmma_kernel<EP_GELU><<<g1, 256, 0, stream>>>(
            ybf, WIDTH, w1 + (size_t)i * MLP_DIM * WIDTH, WIDTH,
            mlp1_b + i * MLP_DIM, nullptr, gbf,
            NTOK, MLP_DIM, WIDTH, NTOK, NTOK, MLP_DIM);

        dim3 g2(NTOK / 16 / 8, WIDTH / 64);
        gemm_wmma_kernel<EP_RESID><<<g2, 256, 0, stream>>>(
            gbf, MLP_DIM, w2 + (size_t)i * WIDTH * MLP_DIM, MLP_DIM,
            mlp2_b + i * WIDTH, hbuf, nullptr,
            NTOK, WIDTH, MLP_DIM, NTOK, NTOK, WIDTH);
    }

    // 6) final LN + sn_linear on latent tokens
    final_kernel<<<Bsz * LTOK, 256, 0, stream>>>(
        hbuf, ln_post_w, ln_post_b, proj_out_w, proj_out_b, sigma_out, (float*)d_out);
}
